// GCN_13520557047989
// MI455X (gfx1250) — matrix operand order repaired and checked by reading, not compiled, not run on hardware
//
#include <hip/hip_runtime.h>

typedef __attribute__((ext_vector_type(16))) _Float16 v16h;
typedef __attribute__((ext_vector_type(8)))  _Float16 v8h;
typedef __attribute__((ext_vector_type(8)))  float    v8f;

#define N_NODES_   500000
#define N_EDGES_   1250000
#define N_FEAT_    128
#define HIDDEN_    64
#define N_GRAPHS_  512
#define N_CLASSES_ 10

// ---------------------------------------------------------------- utilities
__global__ void k_fill(float* __restrict__ p, long long n, float v) {
  long long i = (long long)blockIdx.x * blockDim.x + threadIdx.x;
  if (i < n) p[i] = v;
}

__global__ void k_deg(const int* __restrict__ ei, float* __restrict__ deg) {
  int e = blockIdx.x * blockDim.x + threadIdx.x;
  if (e < N_EDGES_) atomicAdd(&deg[ei[N_EDGES_ + e]], 1.0f);
}

__global__ void k_rsqrt(float* __restrict__ d) {
  int i = blockIdx.x * blockDim.x + threadIdx.x;
  if (i < N_NODES_) d[i] = rsqrtf(d[i]);   // deg >= 1 always (self loops)
}

__global__ void k_cvt_half(const float* __restrict__ s, _Float16* __restrict__ d,
                           long long n) {
  long long i = (long long)blockIdx.x * blockDim.x + threadIdx.x;
  if (i < n) d[i] = (_Float16)s[i];
}

// ------------------------------------------------- weight pre-pack (B frags)
// Packed layout per matrix: [nt][kb][lane][e] halfs, contiguous 32B per lane.
// B-fragment (32x16 f16): lane l holds col n = l&15 ; K = kb*32 + (l>>4)*16 + e.
__global__ void k_pack_w(const float* __restrict__ W0, const float* __restrict__ W1,
                         const float* __restrict__ W2, _Float16* __restrict__ P) {
  int t = blockIdx.x * blockDim.x + threadIdx.x;
  if (t >= 1024) return;                       // 512 (W0) + 256 + 256 combos
  const float* W; int KB; size_t base; int c = t;
  if (c < 512)      { W = W0; KB = 4; base = 0;                 }
  else if (c < 768) { W = W1; KB = 2; base = 8192;  c -= 512;   }
  else              { W = W2; KB = 2; base = 12288; c -= 768;   }
  int lane = c & 31;
  int rest = c >> 5;
  int kb = rest % KB;
  int nt = rest / KB;
  int n     = nt * 16 + (lane & 15);
  int kbase = kb * 32 + (lane >> 4) * 16;
  _Float16* dst = P + base + ((size_t)(nt * KB + kb) * 32 + lane) * 16;
  for (int e = 0; e < 16; ++e)
    dst[e] = (_Float16)W[(size_t)(kbase + e) * HIDDEN_ + n];
}

// ---------------------------------------------------------- WMMA transform
// One wave -> 16 rows x 64 cols of T = Ah @ W.  K/32 steps of
// v_wmma_f32_16x16x32_f16 against 4 accumulators.
template <int K>
__global__ void k_gemm_wmma(const _Float16* __restrict__ Ah,
                            const _Float16* __restrict__ Bp,
                            float* __restrict__ T, int nRows) {
  constexpr int KB = K / 32;
  const int lane = threadIdx.x & 31;
  const int wave = threadIdx.x >> 5;
  const int rowBase = (blockIdx.x * (blockDim.x >> 5) + wave) << 4;
  if (rowBase >= nRows) return;                // nRows % 16 == 0: whole-wave exit
  const int m    = rowBase + (lane & 15);      // A: lanes 0-15 / 16-31 hold M=0..15
  const int aOff = (lane >> 4) * 16;           // A: K sub-block base per lane half
  const _Float16* arow = Ah + (size_t)m * K;

  v8f acc0 = {}, acc1 = {}, acc2 = {}, acc3 = {};
#pragma unroll
  for (int kb = 0; kb < KB; ++kb) {
    // A fragment: elems 0..15 = K kb*32+aOff+e (contiguous, matches B pack)
    v8h lo = *(const v8h*)(arow + kb * 32 + aOff);
    v8h hi = *(const v8h*)(arow + kb * 32 + aOff + 8);
    v16h a = __builtin_shufflevector(lo, hi, 0, 1, 2, 3, 4, 5, 6, 7,
                                     8, 9, 10, 11, 12, 13, 14, 15);
    v16h b0 = *(const v16h*)(Bp + ((size_t)(0 * KB + kb) * 32 + lane) * 16);
    v16h b1 = *(const v16h*)(Bp + ((size_t)(1 * KB + kb) * 32 + lane) * 16);
    v16h b2 = *(const v16h*)(Bp + ((size_t)(2 * KB + kb) * 32 + lane) * 16);
    v16h b3 = *(const v16h*)(Bp + ((size_t)(3 * KB + kb) * 32 + lane) * 16);
    acc0 = __builtin_amdgcn_wmma_f32_16x16x32_f16(false, a, false, b0, (short)0, acc0, false, false);
    acc1 = __builtin_amdgcn_wmma_f32_16x16x32_f16(false, a, false, b1, (short)0, acc1, false, false);
    acc2 = __builtin_amdgcn_wmma_f32_16x16x32_f16(false, a, false, b2, (short)0, acc2, false, false);
    acc3 = __builtin_amdgcn_wmma_f32_16x16x32_f16(false, a, false, b3, (short)0, acc3, false, false);
  }

  // C/D layout: VGPR r -> row r + 8*(lane>>4), col = lane&15
  const int rOff = (lane >> 4) * 8;
  const int col  = lane & 15;
  float* trow = T + (size_t)rowBase * HIDDEN_;
#pragma unroll
  for (int r = 0; r < 8; ++r) {
    size_t base = (size_t)(r + rOff) * HIDDEN_ + col;
    trow[base +  0] = acc0[r];
    trow[base + 16] = acc1[r];
    trow[base + 32] = acc2[r];
    trow[base + 48] = acc3[r];
  }
}

// ------------------------------------------------------- sparse aggregation
__global__ void k_agg_init(const float* __restrict__ T, const float* __restrict__ dinv,
                           float* __restrict__ AGG) {           // self-loop term
  long long idx = (long long)blockIdx.x * blockDim.x + threadIdx.x;
  if (idx >= (long long)N_NODES_ * HIDDEN_) return;
  int i = (int)(idx >> 6);
  float di = dinv[i];
  AGG[idx] = T[idx] * di * di;
}

__global__ void k_agg_edge(const int* __restrict__ ei, const float* __restrict__ T,
                           const float* __restrict__ dinv, float* __restrict__ AGG) {
  long long idx = (long long)blockIdx.x * blockDim.x + threadIdx.x;
  if (idx >= (long long)N_EDGES_ * 16) return;       // 16 threads/edge, 4 ch each
  int e = (int)(idx >> 4);
  int q = (int)(idx & 15);
  int s = ei[e];
  int d = ei[N_EDGES_ + e];
  float w = dinv[s] * dinv[d];
  const float4 v = *(const float4*)(T + (size_t)s * HIDDEN_ + q * 4);
  float* o = AGG + (size_t)d * HIDDEN_ + q * 4;
  atomicAdd(o + 0, v.x * w);
  atomicAdd(o + 1, v.y * w);
  atomicAdd(o + 2, v.z * w);
  atomicAdd(o + 3, v.w * w);
}

__global__ void k_bias_relu(const float* __restrict__ AGG, const float* __restrict__ b,
                            float* __restrict__ F, _Float16* __restrict__ Hh) {
  long long idx = (long long)blockIdx.x * blockDim.x + threadIdx.x;
  if (idx >= (long long)N_NODES_ * HIDDEN_) return;
  int c = (int)(idx & 63);
  float v = fmaxf(AGG[idx] + b[c], 0.0f);
  F[idx]  = v;
  Hh[idx] = (_Float16)v;
}

// ------------------------------------------------------------------ pooling
__global__ void k_pool(const float* __restrict__ F, const int* __restrict__ batch,
                       float* __restrict__ gsum, unsigned* __restrict__ gmax,
                       float* __restrict__ cnt) {
  long long idx = (long long)blockIdx.x * blockDim.x + threadIdx.x;
  if (idx >= (long long)N_NODES_ * HIDDEN_) return;
  int i = (int)(idx >> 6);
  int c = (int)(idx & 63);
  int g = batch[i];
  float v = F[idx];
  atomicAdd(&gsum[g * HIDDEN_ + c], v);
  atomicMax(&gmax[g * HIDDEN_ + c], __float_as_uint(v));  // v >= 0: bit order OK
  if (c == 0) atomicAdd(&cnt[g], 1.0f);
}

__global__ void k_final(const float* __restrict__ gsum, const unsigned* __restrict__ gmax,
                        const float* __restrict__ cnt, const float* __restrict__ Wout,
                        const float* __restrict__ bout, float* __restrict__ out) {
  int t = blockIdx.x * blockDim.x + threadIdx.x;
  if (t >= N_GRAPHS_ * N_CLASSES_) return;
  int g = t / N_CLASSES_, c = t % N_CLASSES_;
  float inv = 1.0f / fmaxf(cnt[g], 1.0f);
  float acc = bout[c];
#pragma unroll 4
  for (int j = 0; j < HIDDEN_; ++j)
    acc += gsum[g * HIDDEN_ + j] * inv * Wout[j * N_CLASSES_ + c];
#pragma unroll 4
  for (int j = 0; j < HIDDEN_; ++j)
    acc += __uint_as_float(gmax[g * HIDDEN_ + j]) * Wout[(HIDDEN_ + j) * N_CLASSES_ + c];
  out[t] = acc;
}

// -------------------------------------------------------------------- launch
extern "C" void kernel_launch(void* const* d_in, const int* in_sizes, int n_in,
                              void* d_out, int out_size, void* d_ws, size_t ws_size,
                              hipStream_t stream) {
  const float* x    = (const float*)d_in[0];
  const int*   ei   = (const int*)d_in[1];
  const int*   bat  = (const int*)d_in[2];
  const float* W0   = (const float*)d_in[3];
  const float* b0   = (const float*)d_in[4];
  const float* W1   = (const float*)d_in[5];
  const float* b1   = (const float*)d_in[6];
  const float* W2   = (const float*)d_in[7];
  const float* b2   = (const float*)d_in[8];
  const float* Wout = (const float*)d_in[9];
  const float* bout = (const float*)d_in[10];
  float* out = (float*)d_out;

  char* ws = (char*)d_ws;
  auto au = [](size_t v) { return (v + 255) & ~(size_t)255; };
  size_t off = 0;
  float*    dinv = (float*)(ws + off);    off = au(off + (size_t)N_NODES_ * 4);
  float*    T    = (float*)(ws + off);    off = au(off + (size_t)N_NODES_ * HIDDEN_ * 4);
  float*    AGG  = (float*)(ws + off);    off = au(off + (size_t)N_NODES_ * HIDDEN_ * 4);
  _Float16* Hh   = (_Float16*)(ws + off); off = au(off + (size_t)N_NODES_ * N_FEAT_ * 2);
  _Float16* Pw   = (_Float16*)(ws + off); off = au(off + (size_t)16384 * 2);
  float*    gsum = (float*)(ws + off);    off = au(off + (size_t)N_GRAPHS_ * HIDDEN_ * 4);
  unsigned* gmax = (unsigned*)(ws + off); off = au(off + (size_t)N_GRAPHS_ * HIDDEN_ * 4);
  float*    cnt  = (float*)(ws + off);    off = au(off + (size_t)N_GRAPHS_ * 4);

  const long long NH  = (long long)N_NODES_ * HIDDEN_;   // 32M
  const long long NF  = (long long)N_NODES_ * N_FEAT_;   // 64M
  const int B = 256;
  const int gNode  = (N_NODES_ + B - 1) / B;
  const int gEdge  = (N_EDGES_ + B - 1) / B;
  const int gNH    = (int)((NH + B - 1) / B);
  const int gNF    = (int)((NF + B - 1) / B);
  const int gE16   = (int)(((long long)N_EDGES_ * 16 + B - 1) / B);
  const int gGemm  = (N_NODES_ + 63) / 64;               // 4 waves/block, 16 rows/wave

  // normalization: deg = 1 (self loop) + in-degree ; dinv = rsqrt(deg)
  k_fill<<<gNode, B, 0, stream>>>(dinv, N_NODES_, 1.0f);
  k_deg<<<gEdge, B, 0, stream>>>(ei, dinv);
  k_rsqrt<<<gNode, B, 0, stream>>>(dinv);

  // weight pack + feature cvt
  k_pack_w<<<4, B, 0, stream>>>(W0, W1, W2, Pw);
  k_cvt_half<<<gNF, B, 0, stream>>>(x, Hh, NF);

  // layer 0 (K=128)
  k_gemm_wmma<128><<<gGemm, 128, 0, stream>>>(Hh, Pw, T, N_NODES_);
  k_agg_init<<<gNH, B, 0, stream>>>(T, dinv, AGG);
  k_agg_edge<<<gE16, B, 0, stream>>>(ei, T, dinv, AGG);
  k_bias_relu<<<gNH, B, 0, stream>>>(AGG, b0, T, Hh);

  // layer 1 (K=64)
  k_gemm_wmma<64><<<gGemm, 128, 0, stream>>>(Hh, Pw + 8192, T, N_NODES_);
  k_agg_init<<<gNH, B, 0, stream>>>(T, dinv, AGG);
  k_agg_edge<<<gE16, B, 0, stream>>>(ei, T, dinv, AGG);
  k_bias_relu<<<gNH, B, 0, stream>>>(AGG, b1, T, Hh);

  // layer 2 (K=64)
  k_gemm_wmma<64><<<gGemm, 128, 0, stream>>>(Hh, Pw + 12288, T, N_NODES_);
  k_agg_init<<<gNH, B, 0, stream>>>(T, dinv, AGG);
  k_agg_edge<<<gE16, B, 0, stream>>>(ei, T, dinv, AGG);
  k_bias_relu<<<gNH, B, 0, stream>>>(AGG, b2, T, Hh);   // T now holds final H (f32)

  // pooling + head
  const long long gsz = (long long)N_GRAPHS_ * HIDDEN_;
  k_fill<<<(int)((gsz + B - 1) / B), B, 0, stream>>>(gsum, gsz, 0.0f);
  k_fill<<<(int)((gsz + B - 1) / B), B, 0, stream>>>((float*)gmax, gsz, 0.0f);
  k_fill<<<(N_GRAPHS_ + B - 1) / B, B, 0, stream>>>(cnt, N_GRAPHS_, 0.0f);
  k_pool<<<gNH, B, 0, stream>>>(T, bat, gsum, gmax, cnt);
  k_final<<<(N_GRAPHS_ * N_CLASSES_ + B - 1) / B, B, 0, stream>>>(gsum, gmax, cnt,
                                                                  Wout, bout, out);
}
